// TransformerBlock_687194767485
// MI455X (gfx1250) — compile-verified, matrix operand order checked by reading, not run on hardware
//
#include <hip/hip_runtime.h>
#include <hip/hip_bf16.h>
#include <math.h>

// ---------------------------------------------------------------------------
// Point Transformer block on gfx1250 (CDNA5).
// All GEMMs: v_wmma_f32_16x16x32_f16 (f16 in, f32 accumulate).
// Tile staging: GLOBAL_LOAD_ASYNC_TO_LDS_B128 + s_wait_asynccnt, double-
// buffered LDS tiles (falls back to sync copies if builtins are absent).
// ---------------------------------------------------------------------------

typedef _Float16 h8  __attribute__((ext_vector_type(8)));
typedef _Float16 h16 __attribute__((ext_vector_type(16)));
typedef float    f8  __attribute__((ext_vector_type(8)));
typedef int      v4i __attribute__((ext_vector_type(4)));

#define SCALE 0.04419417382415922f   // 1/sqrt(512)

#if defined(__has_builtin)
#  if __has_builtin(__builtin_amdgcn_global_load_async_to_lds_b128)
#    define HAVE_ASYNC_LDS 1
#  else
#    define HAVE_ASYNC_LDS 0
#  endif
#else
#  define HAVE_ASYNC_LDS 0
#endif

typedef __attribute__((address_space(1))) v4i* gv4_t;   // global int4*
typedef __attribute__((address_space(3))) v4i* lv4_t;   // LDS int4*

// 16-byte global -> LDS copy (async DMA engine when available)
__device__ inline void cp16(_Float16* lds_dst, const _Float16* g_src) {
#if HAVE_ASYNC_LDS
    __builtin_amdgcn_global_load_async_to_lds_b128((gv4_t)g_src, (lv4_t)lds_dst, 0, 0);
#else
    *(h8*)lds_dst = *(const h8*)g_src;
#endif
}

__device__ inline void wait_async() {
#if HAVE_ASYNC_LDS
#  if __has_builtin(__builtin_amdgcn_s_wait_asynccnt)
    __builtin_amdgcn_s_wait_asynccnt(0);
#  else
    asm volatile("s_wait_asynccnt 0x0" ::: "memory");
#  endif
#endif
}

// Load one 16x32 f16 WMMA fragment (A-row / B-col) from LDS.
// Per CDNA5 ISA 7.12.2: lanes 0-15 hold K 0..7 & 16..23, lanes 16-31 hold
// K 8..15 & 24..31 — two 16B chunks per lane.
__device__ inline h16 load_frag(const _Float16* p, int half) {
    union { h16 v; h8 h[2]; } u;
    u.h[0] = *(const h8*)(p + 8 * half);
    u.h[1] = *(const h8*)(p + 16 + 8 * half);
    return u.v;
}

__device__ inline f8 wmma_f16(h16 a, h16 b, f8 c) {
    return __builtin_amdgcn_wmma_f32_16x16x32_f16(
        false, a, false, b, (short)0, c, false, false);
}

// ---------------------------------------------------------------------------
// Prep kernels: f32 -> f16 convert, and weight transpose W[K,N] -> WT[N,K] f16
// ---------------------------------------------------------------------------
__global__ void cvt_f16(const float* __restrict__ s, _Float16* __restrict__ d, int n) {
    int t = blockIdx.x * 256 + threadIdx.x;
    if (t < n) d[t] = (_Float16)s[t];
}

__global__ void transpose_f16(const float* __restrict__ W, _Float16* __restrict__ WT,
                              int Kd, int N) {
    int t = blockIdx.x * 256 + threadIdx.x;
    if (t >= Kd * N) return;
    int k = t / N, n = t - k * N;
    WT[(size_t)n * Kd + k] = (_Float16)W[t];
}

// ---------------------------------------------------------------------------
// Per-point squared norm
// ---------------------------------------------------------------------------
__global__ void d2_kernel(const float* __restrict__ xyz, float* __restrict__ d2, int n) {
    int t = blockIdx.x * 256 + threadIdx.x;
    if (t < n) {
        float x = xyz[t * 3 + 0], y = xyz[t * 3 + 1], z = xyz[t * 3 + 2];
        d2[t] = x * x + y * y + z * z;
    }
}

// ---------------------------------------------------------------------------
// K=16 nearest neighbors, one wave32 per query. Register-resident sorted
// lists + __shfl_xor wave merge; order matches jax.lax.top_k(-dists, 16).
// ---------------------------------------------------------------------------
__global__ __launch_bounds__(256) void knn_kernel(
    const float* __restrict__ xyz, const float* __restrict__ d2,
    int* __restrict__ knn)
{
    int lane = threadIdx.x & 31, wv = threadIdx.x >> 5;
    int qr = blockIdx.x * 8 + wv;          // b*1024 + n
    int base = (qr >> 10) << 10;
    float qx = xyz[qr * 3 + 0], qy = xyz[qr * 3 + 1], qz = xyz[qr * 3 + 2];
    float d2q = d2[qr];

    float ld[16]; int li[16];
#pragma unroll
    for (int j = 0; j < 16; ++j) { ld[j] = 3.4e38f; li[j] = 0x7fffffff; }

    for (int m = lane; m < 1024; m += 32) {
        const float* p = xyz + (size_t)(base + m) * 3;
        float dot = qx * p[0] + qy * p[1] + qz * p[2];
        float nd = d2q + d2[base + m] - 2.0f * dot;
        int   ni = m;
#pragma unroll
        for (int j = 0; j < 16; ++j) {
            bool sw = (nd < ld[j]) || (nd == ld[j] && ni < li[j]);
            float td = sw ? ld[j] : nd; int ti = sw ? li[j] : ni;
            ld[j] = sw ? nd : ld[j];      li[j] = sw ? ni : li[j];
            nd = td; ni = ti;
        }
    }
    for (int r = 0; r < 16; ++r) {
        float bd = ld[0]; int bi = li[0];
#pragma unroll
        for (int off = 16; off > 0; off >>= 1) {
            float od = __shfl_xor(bd, off, 32);
            int   oi = __shfl_xor(bi, off, 32);
            if (od < bd || (od == bd && oi < bi)) { bd = od; bi = oi; }
        }
        if (ld[0] == bd && li[0] == bi) {
#pragma unroll
            for (int j = 0; j < 15; ++j) { ld[j] = ld[j + 1]; li[j] = li[j + 1]; }
            ld[15] = 3.4e38f; li[15] = 0x7fffffff;
        }
        if (lane == 0) knn[qr * 16 + r] = bi;
    }
}

// ---------------------------------------------------------------------------
// Generic WMMA GEMM (all-f16 operands, async-staged, double-buffered):
//   A:  [M,K]  f16 row-major
//   WT: [N,K]  f16 (weight pre-transposed)
//   outF (f32, bias/relu/residual) and/or outH (f16) = A @ W (+ bias)
// Block tile 128x64, 8 waves x (16 rows x 64 cols), M%128==N%64==K%32==0.
// ---------------------------------------------------------------------------
__global__ __launch_bounds__(256) void gemm_f16wmma(
    const _Float16* __restrict__ A, const _Float16* __restrict__ WT,
    const float* __restrict__ bias, const float* __restrict__ resid,
    float* __restrict__ outF, _Float16* __restrict__ outH,
    int M, int Kdim, int N, int relu)
{
    __shared__ _Float16 At[2][128 * 40];   // padded rows: conflict-free b128
    __shared__ _Float16 Wt[2][64 * 40];

    int tid = threadIdx.x;
    int lane = tid & 31, wv = tid >> 5;
    int half = lane >> 4, l15 = lane & 15;
    int bm = blockIdx.x * 128, bn = blockIdx.y * 64;
    int nk = Kdim >> 5;

    auto stage = [&](int it, int buf) {
        int k0 = it * 32;
        // A tile: 128 rows x 32 halves = 512 x 16B chunks (2 per thread)
#pragma unroll
        for (int i = tid; i < 512; i += 256) {
            int r = i >> 2, s = (i & 3) * 8;
            cp16(&At[buf][r * 40 + s], A + (size_t)(bm + r) * Kdim + k0 + s);
        }
        // W tile: 64 rows x 32 halves = 256 chunks (1 per thread)
        {
            int r = tid >> 2, s = (tid & 3) * 8;
            cp16(&Wt[buf][r * 40 + s], WT + (size_t)(bn + r) * Kdim + k0 + s);
        }
    };

    stage(0, 0);
    f8 acc[4] = {};

    for (int it = 0; it < nk; ++it) {
        wait_async();
        __syncthreads();
        if (it + 1 < nk) stage(it + 1, (it + 1) & 1);

        const _Float16* at = At[it & 1];
        const _Float16* wt = Wt[it & 1];
        h16 a  = load_frag(at + (wv * 16 + l15) * 40, half);
        h16 b0 = load_frag(wt + (0 * 16 + l15) * 40, half);
        h16 b1 = load_frag(wt + (1 * 16 + l15) * 40, half);
        h16 b2 = load_frag(wt + (2 * 16 + l15) * 40, half);
        h16 b3 = load_frag(wt + (3 * 16 + l15) * 40, half);
        acc[0] = wmma_f16(a, b0, acc[0]);
        acc[1] = wmma_f16(a, b1, acc[1]);
        acc[2] = wmma_f16(a, b2, acc[2]);
        acc[3] = wmma_f16(a, b3, acc[3]);
    }

#pragma unroll
    for (int j = 0; j < 4; ++j) {
        int col = bn + j * 16 + l15;
        float bv = bias ? bias[col] : 0.0f;
#pragma unroll
        for (int e = 0; e < 8; ++e) {
            int row = bm + wv * 16 + e + 8 * half;
            float v = acc[j][e] + bv;
            if (relu) v = v > 0.0f ? v : 0.0f;
            if (outH) outH[(size_t)row * N + col] = (_Float16)v;
            if (outF) {
                float o = v;
                if (resid) o += resid[(size_t)row * N + col];
                outF[(size_t)row * N + col] = o;
            }
        }
    }
}

// ---------------------------------------------------------------------------
// Fused attention chain. One block = 64 rows (4 queries x 16 neighbors).
//   stage0: P1  = relu(rel @ Wd1 + bd1)            -> bufA (VALU, K=3)
//   stage1: pos = P1 @ Wd2 + bd2 -> ws f16 ; t = q - k + pos -> bufB
//   stage2: h1  = relu(t @ Wg1 + bg1)              -> bufA
//   stage3: attn_pre = h1 @ Wg2 + bg2              -> d_out attn region (f32)
// Activations LDS-resident across the whole chain (2 x 64x520 f16).
// Weight tiles async-staged from pre-transposed f16 copies, double-buffered.
// ---------------------------------------------------------------------------
template <int MODE>
__device__ inline void chain_gemm(
    int tid, int m0,
    const _Float16* __restrict__ src, _Float16* __restrict__ dst,
    _Float16 (* __restrict__ wt)[128 * 40],
    const _Float16* __restrict__ WT, const float* __restrict__ bias,
    const _Float16* __restrict__ q, const _Float16* __restrict__ kf,
    const int* __restrict__ nbr,
    _Float16* __restrict__ posw, float* __restrict__ attnPre)
{
    int lane = tid & 31, wv = tid >> 5;
    int half = lane >> 4, l15 = lane & 15;
    int strip = wv >> 1, colsub = wv & 1;   // 4 row strips x 2 col halves

    auto stage = [&](int t, int buf) {      // tile t: chunk = t>>4, kstep = t&15
        int ch = t >> 4, ks = t & 15;
#pragma unroll
        for (int i = tid; i < 512; i += 256) {  // 128 rows x 32 halves
            int r = i >> 2, s = (i & 3) * 8;
            cp16(&wt[buf][r * 40 + s],
                 WT + (size_t)(ch * 128 + r) * 512 + ks * 32 + s);
        }
    };

    stage(0, 0);
    f8 acc[4] = {};

    for (int t = 0; t < 64; ++t) {
        int ch = t >> 4, ks = t & 15;
        wait_async();
        __syncthreads();
        if (t + 1 < 64) stage(t + 1, (t + 1) & 1);
        if (ks == 0) { f8 z = {}; acc[0] = z; acc[1] = z; acc[2] = z; acc[3] = z; }

        const _Float16* wb = wt[t & 1];
        h16 a  = load_frag(src + (size_t)(strip * 16 + l15) * 520 + ks * 32, half);
        h16 b0 = load_frag(wb + (colsub * 64 +  0 + l15) * 40, half);
        h16 b1 = load_frag(wb + (colsub * 64 + 16 + l15) * 40, half);
        h16 b2 = load_frag(wb + (colsub * 64 + 32 + l15) * 40, half);
        h16 b3 = load_frag(wb + (colsub * 64 + 48 + l15) * 40, half);
        acc[0] = wmma_f16(a, b0, acc[0]);
        acc[1] = wmma_f16(a, b1, acc[1]);
        acc[2] = wmma_f16(a, b2, acc[2]);
        acc[3] = wmma_f16(a, b3, acc[3]);

        if (ks == 15) {  // epilogue for this 128-col chunk
#pragma unroll
            for (int j = 0; j < 4; ++j) {
                int col = ch * 128 + colsub * 64 + j * 16 + l15;
                float bv = bias[col];
#pragma unroll
                for (int e = 0; e < 8; ++e) {
                    int r = strip * 16 + e + 8 * half;
                    float v = acc[j][e] + bv;
                    if (MODE == 1) {
                        size_t g = (size_t)(m0 + r) * 512 + col;
                        posw[g] = (_Float16)v;
                        int qrow = (m0 + r) >> 4;
                        float tv = (float)q[(size_t)qrow * 512 + col]
                                 - (float)kf[(size_t)nbr[r] * 512 + col] + v;
                        dst[(size_t)r * 520 + col] = (_Float16)tv;
                    } else if (MODE == 2) {
                        dst[(size_t)r * 520 + col] = (_Float16)(v > 0.0f ? v : 0.0f);
                    } else {
                        attnPre[(size_t)(m0 + r) * 512 + col] = v;
                    }
                }
            }
        }
    }
    __syncthreads();
}

__global__ __launch_bounds__(256) void attn_chain(
    const float* __restrict__ xyz,
    const _Float16* __restrict__ q, const _Float16* __restrict__ kf,
    const int* __restrict__ knn,
    const float* __restrict__ Wd1, const float* __restrict__ bd1,
    const _Float16* __restrict__ Wd2T, const float* __restrict__ bd2,
    const _Float16* __restrict__ Wg1T, const float* __restrict__ bg1,
    const _Float16* __restrict__ Wg2T, const float* __restrict__ bg2,
    _Float16* __restrict__ posw, float* __restrict__ attnPre)
{
    __shared__ _Float16 bufA[64 * 520];    // 66,560 B
    __shared__ _Float16 bufB[64 * 520];    // 66,560 B
    __shared__ _Float16 wt[2][128 * 40];   // 20,480 B (double-buffered)
    __shared__ float rel[64][3];
    __shared__ int   nbr[64];

    int tid = threadIdx.x;
    int m0 = blockIdx.x * 64;

    if (tid < 64) {
        int m = m0 + tid;
        int qr = m >> 4, kk = m & 15;
        int b = qr >> 10;
        int nb = (b << 10) + knn[qr * 16 + kk];
        nbr[tid] = nb;
        rel[tid][0] = xyz[qr * 3 + 0] - xyz[nb * 3 + 0];
        rel[tid][1] = xyz[qr * 3 + 1] - xyz[nb * 3 + 1];
        rel[tid][2] = xyz[qr * 3 + 2] - xyz[nb * 3 + 2];
    }
    __syncthreads();

    // stage 0: P1 = relu(rel @ Wd1 + bd1), K=3 -> VALU into bufA
    for (int i = tid; i < 64 * 512; i += 256) {
        int r = i >> 9, c = i & 511;
        float v = bd1[c] + rel[r][0] * Wd1[c]
                         + rel[r][1] * Wd1[512 + c]
                         + rel[r][2] * Wd1[1024 + c];
        bufA[(size_t)r * 520 + c] = (_Float16)(v > 0.0f ? v : 0.0f);
    }
    __syncthreads();

    chain_gemm<1>(tid, m0, bufA, bufB, wt, Wd2T, bd2, q, kf, nbr, posw, attnPre);
    chain_gemm<2>(tid, m0, bufB, bufA, wt, Wg1T, bg1, q, kf, nbr, posw, attnPre);
    chain_gemm<3>(tid, m0, bufA, bufB, wt, Wg2T, bg2, q, kf, nbr, posw, attnPre);
}

// ---------------------------------------------------------------------------
// Softmax over the 16 neighbors per (query, channel), in place on d_out.
// ---------------------------------------------------------------------------
__global__ void softmax_k(float* __restrict__ attn) {
    int t = blockIdx.x * 256 + threadIdx.x;        // qr*512 + c
    if (t >= 4096 * 512) return;
    int qr = t >> 9, c = t & 511;
    size_t base = (size_t)qr * 16 * 512 + c;
    float vals[16];
    float mx = -3.4e38f;
#pragma unroll
    for (int k = 0; k < 16; ++k) {
        vals[k] = attn[base + (size_t)k * 512] * SCALE;
        mx = fmaxf(mx, vals[k]);
    }
    float s = 0.0f;
#pragma unroll
    for (int k = 0; k < 16; ++k) { vals[k] = expf(vals[k] - mx); s += vals[k]; }
    float inv = 1.0f / s;
#pragma unroll
    for (int k = 0; k < 16; ++k) attn[base + (size_t)k * 512] = vals[k] * inv;
}

// ---------------------------------------------------------------------------
// y[qr, c] = sum_k attn[qr,k,c] * (v[gather(qr,k), c] + pos[qr,k,c]) -> f16
// ---------------------------------------------------------------------------
__global__ void reduce_k(
    const float* __restrict__ attn, const _Float16* __restrict__ vh,
    const _Float16* __restrict__ posw, const int* __restrict__ knn,
    _Float16* __restrict__ yh)
{
    int t = blockIdx.x * 256 + threadIdx.x;
    if (t >= 4096 * 512) return;
    int qr = t >> 9, c = t & 511;
    int b = qr >> 10;
    size_t base = (size_t)qr * 16 * 512 + c;
    float s = 0.0f;
#pragma unroll
    for (int k = 0; k < 16; ++k) {
        int nb = (b << 10) + knn[qr * 16 + k];
        float vv = (float)vh[(size_t)nb * 512 + c] + (float)posw[base + (size_t)k * 512];
        s += attn[base + (size_t)k * 512] * vv;
    }
    yh[(size_t)qr * 512 + c] = (_Float16)s;
}

// ---------------------------------------------------------------------------
// Host launcher
// ---------------------------------------------------------------------------
extern "C" void kernel_launch(void* const* d_in, const int* in_sizes, int n_in,
                              void* d_out, int out_size, void* d_ws, size_t ws_size,
                              hipStream_t stream) {
    (void)in_sizes; (void)n_in; (void)out_size; (void)ws_size;
    const float* xyz      = (const float*)d_in[0];
    const float* features = (const float*)d_in[1];
    const float* W1  = (const float*)d_in[2];
    const float* b1  = (const float*)d_in[3];
    const float* W2  = (const float*)d_in[4];
    const float* b2  = (const float*)d_in[5];
    const float* Wd1 = (const float*)d_in[6];
    const float* bd1 = (const float*)d_in[7];
    const float* Wd2 = (const float*)d_in[8];
    const float* bd2 = (const float*)d_in[9];
    const float* Wg1 = (const float*)d_in[10];
    const float* bg1 = (const float*)d_in[11];
    const float* Wg2 = (const float*)d_in[12];
    const float* bg2 = (const float*)d_in[13];
    const float* Wq  = (const float*)d_in[14];
    const float* Wk  = (const float*)d_in[15];
    const float* Wv  = (const float*)d_in[16];

    float* res_out  = (float*)d_out;                       // [4096, 256]
    float* attn_out = (float*)d_out + (size_t)4096 * 256;  // [4096*16, 512]

    // workspace (~90 MB)
    char* w = (char*)d_ws;
    size_t off = 0;
    auto take = [&](size_t bytes) {
        char* p = w + off; off = (off + bytes + 255) & ~(size_t)255; return p;
    };
    float*    d2w   = (float*)take((size_t)4096 * 4);
    int*      knnw  = (int*)  take((size_t)4096 * 16 * 4);
    _Float16* featH = (_Float16*)take((size_t)4096 * 256 * 2);
    _Float16* xh    = (_Float16*)take((size_t)4096 * 512 * 2);
    _Float16* qh    = (_Float16*)take((size_t)4096 * 512 * 2);
    _Float16* kh    = (_Float16*)take((size_t)4096 * 512 * 2);
    _Float16* vh    = (_Float16*)take((size_t)4096 * 512 * 2);
    _Float16* posw  = (_Float16*)take((size_t)65536 * 512 * 2);
    _Float16* yh    = (_Float16*)take((size_t)4096 * 512 * 2);
    _Float16* W1T   = (_Float16*)take((size_t)512 * 256 * 2);
    _Float16* WqT   = (_Float16*)take((size_t)512 * 512 * 2);
    _Float16* WkT   = (_Float16*)take((size_t)512 * 512 * 2);
    _Float16* WvT   = (_Float16*)take((size_t)512 * 512 * 2);
    _Float16* Wd2T  = (_Float16*)take((size_t)512 * 512 * 2);
    _Float16* Wg1T  = (_Float16*)take((size_t)512 * 512 * 2);
    _Float16* Wg2T  = (_Float16*)take((size_t)512 * 512 * 2);
    _Float16* W2T   = (_Float16*)take((size_t)256 * 512 * 2);

    // 0) precision/layout prep
    cvt_f16<<<4096, 256, 0, stream>>>(features, featH, 4096 * 256);
    transpose_f16<<< 512, 256, 0, stream>>>(W1,  W1T,  256, 512);
    transpose_f16<<<1024, 256, 0, stream>>>(Wq,  WqT,  512, 512);
    transpose_f16<<<1024, 256, 0, stream>>>(Wk,  WkT,  512, 512);
    transpose_f16<<<1024, 256, 0, stream>>>(Wv,  WvT,  512, 512);
    transpose_f16<<<1024, 256, 0, stream>>>(Wd2, Wd2T, 512, 512);
    transpose_f16<<<1024, 256, 0, stream>>>(Wg1, Wg1T, 512, 512);
    transpose_f16<<<1024, 256, 0, stream>>>(Wg2, Wg2T, 512, 512);
    transpose_f16<<< 512, 256, 0, stream>>>(W2,  W2T,  512, 256);

    // 1) KNN
    d2_kernel <<<16,  256, 0, stream>>>(xyz, d2w, 4096);
    knn_kernel<<<512, 256, 0, stream>>>(xyz, d2w, knnw);

    // 2) x = features @ W1 + b1 ; q/k/v = x @ W{q,k,v}   (f16 outputs)
    gemm_f16wmma<<<dim3(32, 8), 256, 0, stream>>>(featH, W1T, b1, nullptr,
                                                  nullptr, xh, 4096, 256, 512, 0);
    gemm_f16wmma<<<dim3(32, 8), 256, 0, stream>>>(xh, WqT, nullptr, nullptr,
                                                  nullptr, qh, 4096, 512, 512, 0);
    gemm_f16wmma<<<dim3(32, 8), 256, 0, stream>>>(xh, WkT, nullptr, nullptr,
                                                  nullptr, kh, 4096, 512, 512, 0);
    gemm_f16wmma<<<dim3(32, 8), 256, 0, stream>>>(xh, WvT, nullptr, nullptr,
                                                  nullptr, vh, 4096, 512, 512, 0);

    // 3) fused pos_enc -> t -> gate1 -> gate2 chain (LDS-resident)
    attn_chain<<<1024, 256, 0, stream>>>(xyz, qh, kh, knnw,
                                         Wd1, bd1, Wd2T, bd2,
                                         Wg1T, bg1, Wg2T, bg2,
                                         posw, attn_out);

    // 4) softmax over neighbors, in place in d_out
    softmax_k<<<8192, 256, 0, stream>>>(attn_out);

    // 5) y = sum_k attn * (v + pos)  (f16 out)
    reduce_k<<<8192, 256, 0, stream>>>(attn_out, vh, posw, knnw, yh);

    // 6) res = y @ W2 + b2 + features
    gemm_f16wmma<<<dim3(32, 4), 256, 0, stream>>>(yh, W2T, b2, features,
                                                  res_out, nullptr, 4096, 512, 256, 0);
}
